// OptimizedDecoderLayer_61770219651463
// MI455X (gfx1250) — matrix operand order repaired and checked by reading, not compiled, not run on hardware
//
#include <hip/hip_runtime.h>
#include <cstdint>

#define EPS 1e-5f

// ---------------- CDNA5 WMMA types / helpers ----------------
typedef __attribute__((ext_vector_type(16))) __bf16 v16bf;
typedef __attribute__((ext_vector_type(8)))  float  v8f;

__device__ __forceinline__ v8f v8f_zero() {
  v8f z;
#pragma unroll
  for (int i = 0; i < 8; ++i) z[i] = 0.0f;
  return z;
}

__device__ __forceinline__ v8f wmma_bf16(v16bf a, v16bf b, v8f c) {
  // D = A(16x32) * B(32x16) + C(16x16), f32 accumulate
  return __builtin_amdgcn_wmma_f32_16x16x32_bf16(false, a, false, b, (short)0, c,
                                                 false, false);
}

// ---------------- CDNA5 async global->LDS copy + split waits ----------------
__device__ __forceinline__ void async_g2l_b128(void* lds, const void* gptr) {
  uint32_t loff = (uint32_t)(uintptr_t)lds;  // flat addr low 32 bits == LDS offset
  asm volatile("global_load_async_to_lds_b128 %0, %1, off"
               :: "v"(loff), "v"(gptr) : "memory");
}
template <int N>
__device__ __forceinline__ void wait_async() {
  asm volatile("s_wait_asynccnt %0" :: "i"(N) : "memory");
}

// ---------------- DPP 16-lane reductions (VALU, keeps LDS unit free) --------
// ctrl: 0xB1 = quad_perm xor1, 0x4E = quad_perm xor2,
//       0x141 = row_half_mirror (lane^7), 0x140 = row_mirror (lane^15)
template <int CTRL>
__device__ __forceinline__ float dpp_mov(float x) {
  return __int_as_float(
      __builtin_amdgcn_update_dpp(0, __float_as_int(x), CTRL, 0xf, 0xf, true));
}
__device__ __forceinline__ float rowmax16(float x) {
  x = fmaxf(x, dpp_mov<0xB1>(x));
  x = fmaxf(x, dpp_mov<0x4E>(x));
  x = fmaxf(x, dpp_mov<0x141>(x));
  x = fmaxf(x, dpp_mov<0x140>(x));
  return x;
}
__device__ __forceinline__ float rowsum16(float x) {
  x += dpp_mov<0xB1>(x);
  x += dpp_mov<0x4E>(x);
  x += dpp_mov<0x141>(x);
  x += dpp_mov<0x140>(x);
  return x;
}

// ---------------- fragment builders (ISA 7.12.2 layouts) ----------------
// A 16x32 bf16: lane l: M = l%16; kbase=(l<16)?0:8; elem e -> K = kbase+(e&7)+16*(e>>3)
__device__ __forceinline__ v16bf make_a_frag_f32(const float* tile, int row0, int ld,
                                                 float scale) {
  const int l = threadIdx.x & 31;
  const int m = l & 15;
  const int kb = (l < 16) ? 0 : 8;
  const float* r = tile + (size_t)(row0 + m) * ld;
  v16bf a;
#pragma unroll
  for (int e = 0; e < 16; ++e) {
    int k = kb + (e & 7) + ((e >> 3) << 4);
    a[e] = (__bf16)(r[k] * scale);
  }
  return a;
}

__device__ __forceinline__ v16bf make_a_frag_bf16(const __bf16* tile, int ld) {
  const int l = threadIdx.x & 31;
  const int m = l & 15;
  const int kb = (l < 16) ? 0 : 8;
  const __bf16* r = tile + m * ld;
  v16bf a;
#pragma unroll
  for (int e = 0; e < 16; ++e) a[e] = r[kb + (e & 7) + ((e >> 3) << 4)];
  return a;
}

// B 32x16 bf16 from an [N][K] tile: lane l: N=l%16; kbase=(l<16)?0:8; K=kbase+(e&7)+16*(e>>3)
__device__ __forceinline__ v16bf make_b_frag_nk(const float* tile, int n0, int ld) {
  const int l = threadIdx.x & 31;
  const int n = l & 15;
  const int kb = (l < 16) ? 0 : 8;
  const float* r = tile + (size_t)(n0 + n) * ld + kb;
  v16bf b;
#pragma unroll
  for (int e = 0; e < 16; ++e) b[e] = (__bf16)r[(e & 7) + ((e >> 3) << 4)];
  return b;
}

// B 32x16 bf16 from a [K][N] tile (V chunk: rows=keys, cols=dh)
__device__ __forceinline__ v16bf make_b_frag_kn(const float* tile, int n0, int ld) {
  const int l = threadIdx.x & 31;
  const int n = (l & 15) + n0;
  const int kb = (l < 16) ? 0 : 8;
  v16bf b;
#pragma unroll
  for (int e = 0; e < 16; ++e) b[e] = (__bf16)tile[(size_t)(kb + (e & 7) + ((e >> 3) << 4)) * ld + n];
  return b;
}

// ---------------- K1: pair2heads bias: LN(p=64) + Linear(64->8) + mask -------
__global__ __launch_bounds__(256) void pair_bias_kernel(
    const float* __restrict__ pf, const float* __restrict__ mask,
    const float* __restrict__ g, const float* __restrict__ beta,
    const float* __restrict__ pw, const float* __restrict__ pb,
    float* __restrict__ out) {
  __shared__ float sw[8 * 64];
  __shared__ float sb[8];
  __shared__ float sg[64];
  __shared__ float sbe[64];
  const int tid = threadIdx.x;
  for (int i = tid; i < 512; i += 256) sw[i] = pw[i];
  if (tid < 8) sb[tid] = pb[tid];
  if (tid >= 32 && tid < 96) sg[tid - 32] = g[tid - 32];
  if (tid >= 96 && tid < 160) sbe[tid - 96] = beta[tid - 96];
  __syncthreads();

  const size_t gidx = (size_t)blockIdx.x * 256 + tid;  // over 2*512*512
  const int j = (int)(gidx & 511);
  const size_t bi = gidx >> 9;
  const int i = (int)(bi & 511);
  const int b = (int)(bi >> 9);
  const float* x = pf + gidx * 64;

  float v[64];
  float s1 = 0.f, s2 = 0.f;
#pragma unroll
  for (int p = 0; p < 64; ++p) { float t = x[p]; v[p] = t; s1 += t; s2 += t * t; }
  const float mu = s1 * (1.f / 64.f);
  const float var = s2 * (1.f / 64.f) - mu * mu;
  const float rs = rsqrtf(var + EPS);
#pragma unroll
  for (int p = 0; p < 64; ++p) v[p] = (v[p] - mu) * rs * sg[p] + sbe[p];

  const float mk = mask[(size_t)i * 512 + j];
#pragma unroll
  for (int h = 0; h < 8; ++h) {
    float acc = sb[h];
    const float* wr = &sw[h * 64];
#pragma unroll
    for (int p = 0; p < 64; ++p) acc += v[p] * wr[p];
    out[((((size_t)b * 8 + h) * 512) + i) * 512 + j] = acc + mk;
  }
}

// ---------------- K2: double-buffered WMMA GEMM ------------------------------
// C[M,N] = act(A[M,K] @ W[N,K]^T + bias); split-A fuses [x; memory] concat.
// Block 256 = 8 waves, tile 128x64, waves 4(M) x 2(N), each wave 32x32.
// LDS rows padded to 36 floats: keeps b128 stores 16B aligned, bank-conflict free.
#define LDA 36

__device__ __forceinline__ void stage_gemm_tile(
    float* At, float* Wt, const float* A, const float* A2, int k1, int k2,
    const float* W, int m0, int n0, int K, int kk, int tid) {
#pragma unroll
  for (int it = 0; it < 4; ++it) {
    int s = tid + it * 256;
    int row = s >> 3;
    int cg = (s & 7) << 2;
    int gc = kk + cg;
    const float* src = (gc < k1) ? (A + (size_t)(m0 + row) * k1 + gc)
                                 : (A2 + (size_t)(m0 + row) * k2 + (gc - k1));
    async_g2l_b128(&At[row * LDA + cg], src);
  }
#pragma unroll
  for (int it = 0; it < 2; ++it) {
    int s = tid + it * 256;
    int row = s >> 3;
    int cg = (s & 7) << 2;
    async_g2l_b128(&Wt[row * LDA + cg], W + (size_t)(n0 + row) * K + kk + cg);
  }
}

template <int ACT>
__global__ __launch_bounds__(256) void gemm_kernel(
    const float* __restrict__ A, const float* __restrict__ A2, int k1,
    const float* __restrict__ W, const float* __restrict__ bias,
    float* __restrict__ C, int M, int N, int K) {
  __shared__ float Atile[2][128 * LDA];
  __shared__ float Wtile[2][64 * LDA];
  const int tid = threadIdx.x;
  const int wave = tid >> 5;
  const int l = tid & 31;
  const int wm = wave >> 1;  // 0..3
  const int wn = wave & 1;   // 0..1
  const int m0 = blockIdx.x * 128;
  const int n0 = blockIdx.y * 64;
  const int k2 = K - k1;

  v8f acc[2][2];
#pragma unroll
  for (int i = 0; i < 2; ++i)
#pragma unroll
    for (int j = 0; j < 2; ++j) acc[i][j] = v8f_zero();

  const int nt = K >> 5;
  stage_gemm_tile(Atile[0], Wtile[0], A, A2, k1, k2, W, m0, n0, K, 0, tid);
  for (int t = 0; t < nt; ++t) {
    if (t + 1 < nt) {
      stage_gemm_tile(Atile[(t + 1) & 1], Wtile[(t + 1) & 1], A, A2, k1, k2, W,
                      m0, n0, K, (t + 1) << 5, tid);
      wait_async<6>();  // tile t's 6 asyncs retired (in-order); t+1 in flight
    } else {
      wait_async<0>();
    }
    __syncthreads();

    const float* At = Atile[t & 1];
    const float* Wt = Wtile[t & 1];
#pragma unroll
    for (int i = 0; i < 2; ++i) {
      v16bf a = make_a_frag_f32(At, wm * 32 + i * 16, LDA, 1.0f);
#pragma unroll
      for (int j = 0; j < 2; ++j) {
        v16bf b = make_b_frag_nk(Wt, wn * 32 + j * 16, LDA);
        acc[i][j] = wmma_bf16(a, b, acc[i][j]);
      }
    }
    __syncthreads();  // everyone done with buf[t&1] before t+2 overwrites it
  }

  // epilogue: C-frag elem r -> M = r + 8*(l>=16), N = l%16
  const int mh = (l >> 4) * 8;
  const int nl = l & 15;
#pragma unroll
  for (int i = 0; i < 2; ++i) {
#pragma unroll
    for (int j = 0; j < 2; ++j) {
      const int gn = n0 + wn * 32 + j * 16 + nl;
      const float bv = bias[gn];
#pragma unroll
      for (int r = 0; r < 8; ++r) {
        float v = acc[i][j][r] + bv;
        if (ACT == 1) v = fmaxf(v, 0.f);
        C[(size_t)(m0 + wm * 32 + i * 16 + mh + r) * N + gn] = v;
      }
    }
  }
}

// ---------------- K3: flash attention, double-buffered K/V ------------------
// blockIdx.x = bh*4 + qc ; bh = bnr*8 + h ; 8 waves, each owns 16 query rows.
#define LDP 40  // Pbuf row stride (bf16 elems): conflict-free fragment reads

__global__ __launch_bounds__(256) void attn_kernel(
    const float* __restrict__ qkv, const float* __restrict__ bias,
    float* __restrict__ ctx) {
  const int L = 512, D3 = 768;
  const int bh = blockIdx.x >> 2;
  const int qc = blockIdx.x & 3;
  const int bnr = bh >> 3;
  const int h = bh & 7;
  const int b2 = bnr >> 3;  // / NR
  const int wave = threadIdx.x >> 5;
  const int l = threadIdx.x & 31;
  const int qbase = qc * 128 + wave * 16;
  const size_t rowbase = (size_t)bnr * L;

  __shared__ float Ktile[2][32 * LDA];
  __shared__ float Vtile[2][32 * LDA];
  __shared__ __bf16 Pbuf[8][16 * LDP];

  // q fragment, pre-scaled by 1/sqrt(dh)
  const float fscale = 0.17677669529663687f;
  v16bf qfrag;
  {
    const int m = l & 15;
    const int kb = (l < 16) ? 0 : 8;
    const float* qr = qkv + (rowbase + qbase + m) * D3 + h * 32;
#pragma unroll
    for (int e = 0; e < 16; ++e)
      qfrag[e] = (__bf16)(qr[kb + (e & 7) + ((e >> 3) << 4)] * fscale);
  }

  const float* Kbase = qkv + rowbase * D3 + 256 + h * 32;
  const float* Vbase = qkv + rowbase * D3 + 512 + h * 32;
  const float* brow = bias + (((size_t)b2 * 8 + h) * L) * L;

  v8f av[2];
  av[0] = v8f_zero();
  av[1] = v8f_zero();
  float mrow[8], lrow[8];
#pragma unroll
  for (int r = 0; r < 8; ++r) { mrow[r] = -1e30f; lrow[r] = 0.f; }

  const int mh = (l >> 4) * 8;
  const int nl = l & 15;
  const int srow = threadIdx.x >> 3;          // staging row 0..31
  const int scg = (threadIdx.x & 7) << 2;     // staging col group

  // prologue: stage chunk 0 (2 async instrs per wave)
  async_g2l_b128(&Ktile[0][srow * LDA + scg], Kbase + (size_t)srow * D3 + scg);
  async_g2l_b128(&Vtile[0][srow * LDA + scg], Vbase + (size_t)srow * D3 + scg);

  for (int c = 0; c < 16; ++c) {
    if (c + 1 < 16) {
      const int nk = (c + 1) * 32;
      async_g2l_b128(&Ktile[(c + 1) & 1][srow * LDA + scg],
                     Kbase + (size_t)(nk + srow) * D3 + scg);
      async_g2l_b128(&Vtile[(c + 1) & 1][srow * LDA + scg],
                     Vbase + (size_t)(nk + srow) * D3 + scg);
      wait_async<2>();  // chunk c resident; chunk c+1 in flight
    } else {
      wait_async<0>();
    }
    __syncthreads();

    const float* Kt = Ktile[c & 1];
    const float* Vt = Vtile[c & 1];
    const int key0 = c * 32;

    // S(16x32) = q @ K^T as two 16x16 fragments
    v8f s0 = wmma_bf16(qfrag, make_b_frag_nk(Kt, 0, LDA), v8f_zero());
    v8f s1 = wmma_bf16(qfrag, make_b_frag_nk(Kt, 16, LDA), v8f_zero());

#pragma unroll
    for (int r = 0; r < 8; ++r) {
      const int qr = qbase + mh + r;
      s0[r] += brow[(size_t)qr * L + key0 + nl];
      s1[r] += brow[(size_t)qr * L + key0 + 16 + nl];
      const float mx = rowmax16(fmaxf(s0[r], s1[r]));
      const float mnew = fmaxf(mrow[r], mx);
      const float corr = __expf(mrow[r] - mnew);
      const float p0 = __expf(s0[r] - mnew);
      const float p1 = __expf(s1[r] - mnew);
      const float ps = rowsum16(p0 + p1);
      lrow[r] = lrow[r] * corr + ps;
      av[0][r] *= corr;
      av[1][r] *= corr;
      mrow[r] = mnew;
      Pbuf[wave][(mh + r) * LDP + nl] = (__bf16)p0;
      Pbuf[wave][(mh + r) * LDP + 16 + nl] = (__bf16)p1;
    }

    // ctx += P(16x32) @ V(32x32)  (wave-private Pbuf -> A fragment)
    v16bf pfr = make_a_frag_bf16(&Pbuf[wave][0], LDP);
    av[0] = wmma_bf16(pfr, make_b_frag_kn(Vt, 0, LDA), av[0]);
    av[1] = wmma_bf16(pfr, make_b_frag_kn(Vt, 16, LDA), av[1]);
    __syncthreads();
  }

#pragma unroll
  for (int j = 0; j < 2; ++j)
#pragma unroll
    for (int r = 0; r < 8; ++r)
      ctx[(rowbase + qbase + mh + r) * 256 + h * 32 + j * 16 + nl] =
          av[j][r] / lrow[r];
}

// ---------------- K4/K5: fused residual (+gated cc) + LayerNorm over d=256 ---
// MODE 0: y = LN(a + b) ; MODE 1: y = LN(a + sigmoid(b)*c). One wave per row.
template <int MODE>
__global__ __launch_bounds__(256) void addln_kernel(
    const float* __restrict__ a, const float* __restrict__ b,
    const float* __restrict__ c, const float* __restrict__ g,
    const float* __restrict__ beta, float* __restrict__ out, int nrows) {
  const int wave = threadIdx.x >> 5;
  const int l = threadIdx.x & 31;
  const int row = blockIdx.x * 8 + wave;
  if (row >= nrows) return;
  const float* ar = a + (size_t)row * 256;
  const float* br = b + (size_t)row * 256;
  const float* cr = (MODE == 1) ? (c + (size_t)row * 256) : nullptr;

  float v[8];
  float s1 = 0.f, s2 = 0.f;
#pragma unroll
  for (int i = 0; i < 8; ++i) {
    const int idx = l + i * 32;
    float x = ar[idx];
    if (MODE == 0) {
      x += br[idx];
    } else {
      const float gt = 1.f / (1.f + __expf(-br[idx]));
      x += gt * cr[idx];
    }
    v[i] = x;
    s1 += x;
    s2 += x * x;
  }
#pragma unroll
  for (int off = 1; off < 32; off <<= 1) {
    s1 += __shfl_xor(s1, off, 32);
    s2 += __shfl_xor(s2, off, 32);
  }
  const float mu = s1 * (1.f / 256.f);
  const float var = s2 * (1.f / 256.f) - mu * mu;
  const float rs = rsqrtf(var + EPS);
  float* orow = out + (size_t)row * 256;
#pragma unroll
  for (int i = 0; i < 8; ++i) {
    const int idx = l + i * 32;
    orow[idx] = (v[i] - mu) * rs * g[idx] + beta[idx];
  }
}

// ---------------- host orchestration ----------------
extern "C" void kernel_launch(void* const* d_in, const int* in_sizes, int n_in,
                              void* d_out, int out_size, void* d_ws, size_t ws_size,
                              hipStream_t stream) {
  const float* tgt      = (const float*)d_in[0];
  const float* memory   = (const float*)d_in[1];
  const float* pf       = (const float*)d_in[2];
  const float* mask     = (const float*)d_in[3];
  const float* p_ln_g   = (const float*)d_in[4];
  const float* p_ln_b   = (const float*)d_in[5];
  const float* p_w      = (const float*)d_in[6];
  const float* p_b      = (const float*)d_in[7];
  const float* in_w     = (const float*)d_in[8];
  const float* in_b     = (const float*)d_in[9];
  const float* out_w    = (const float*)d_in[10];
  const float* out_b    = (const float*)d_in[11];
  const float* cc_lin_w = (const float*)d_in[12];
  const float* cc_lin_b = (const float*)d_in[13];
  const float* cc_gate_w= (const float*)d_in[14];
  const float* cc_gate_b= (const float*)d_in[15];
  const float* ff_w1    = (const float*)d_in[16];
  const float* ff_b1    = (const float*)d_in[17];
  const float* ff_w2    = (const float*)d_in[18];
  const float* ff_b2    = (const float*)d_in[19];
  const float* n1_g     = (const float*)d_in[20];
  const float* n1_b     = (const float*)d_in[21];
  const float* n2_g     = (const float*)d_in[22];
  const float* n2_b     = (const float*)d_in[23];
  const float* n3_g     = (const float*)d_in[24];
  const float* n3_b     = (const float*)d_in[25];

  const int M = 16 * 512;  // 8192 rows
  float* ws = (float*)d_ws;
  // workspace layout (floats), liveness-aliased (75.5 MB total):
  float* bias    = ws;                   // 4,194,304  (later: gate | lin)
  float* big     = bias + 4194304;       // 8,388,608  (qkv, later ff1)
  float* ctxbuf  = big + 8388608;        // 2,097,152  (ctx, later ff2)
  float* attnout = ctxbuf + 2097152;     // 2,097,152  (attnout, later x2)
  float* x1      = attnout + 2097152;    // 2,097,152

  // 1) pairwise bias: [2,8,512,512]
  pair_bias_kernel<<<2048, 256, 0, stream>>>(pf, mask, p_ln_g, p_ln_b, p_w, p_b, bias);

  // 2) qkv = tgt @ in_w^T + in_b : [8192,768]
  float* qkv = big;
  gemm_kernel<0><<<dim3(M / 128, 768 / 64), 256, 0, stream>>>(
      tgt, tgt, 256, in_w, in_b, qkv, M, 768, 256);

  // 3) flash attention -> ctx [8192,256]
  attn_kernel<<<512, 256, 0, stream>>>(qkv, bias, ctxbuf);

  // 4) out projection
  gemm_kernel<0><<<dim3(M / 128, 256 / 64), 256, 0, stream>>>(
      ctxbuf, ctxbuf, 256, out_w, out_b, attnout, M, 256, 256);

  // 5) x1 = LN(tgt + attnout)
  addln_kernel<0><<<M / 8, 256, 0, stream>>>(tgt, attnout, nullptr, n1_g, n1_b, x1, M);

  // 6) cross-concat gate / lin on [x1; memory] (K=512 split at 256)
  float* gate = bias;            // bias dead after attention
  float* lin  = bias + 2097152;
  gemm_kernel<0><<<dim3(M / 128, 256 / 64), 256, 0, stream>>>(
      x1, memory, 256, cc_gate_w, cc_gate_b, gate, M, 256, 512);
  gemm_kernel<0><<<dim3(M / 128, 256 / 64), 256, 0, stream>>>(
      x1, memory, 256, cc_lin_w, cc_lin_b, lin, M, 256, 512);

  // 7) x2 = LN(x1 + sigmoid(gate)*lin)
  float* x2 = attnout;  // attnout dead
  addln_kernel<1><<<M / 8, 256, 0, stream>>>(x1, gate, lin, n2_g, n2_b, x2, M);

  // 8) FFN
  float* ff1 = big;     // qkv dead
  gemm_kernel<1><<<dim3(M / 128, 1024 / 64), 256, 0, stream>>>(
      x2, x2, 256, ff_w1, ff_b1, ff1, M, 1024, 256);
  float* ff2 = ctxbuf;  // ctx dead
  gemm_kernel<0><<<dim3(M / 128, 256 / 64), 256, 0, stream>>>(
      ff1, ff1, 1024, ff_w2, ff_b2, ff2, M, 256, 1024);

  // 9) out = LN(x2 + ff2)
  addln_kernel<0><<<M / 8, 256, 0, stream>>>(x2, ff2, nullptr, n3_g, n3_b,
                                             (float*)d_out, M);
}
